// PNAConv_EdgeAttrib_44281112822530
// MI455X (gfx1250) — compile-verified, matrix operand order checked by reading, not run on hardware
//
#include <hip/hip_runtime.h>
#include <hip/hip_bf16.h>
#include <math.h>
#include <stdint.h>

// ---------------------------------------------------------------------------
// PNAConv (3 layers, 2 towers, 5 agg x 5 scalers) + MLP head on gfx1250.
// f16 WMMA (f32 accumulate) for every dense GEMM; weights pre-packed into
// WMMA B-fragment layout; GEMM-only intermediates stored f16 with padded
// rows (A fragment = two global_load_b128); each block computes 3 column
// tiles per A fragment (3 wmma per A load).
// ---------------------------------------------------------------------------

typedef __attribute__((ext_vector_type(16))) _Float16 v16h;
typedef __attribute__((ext_vector_type(8)))  _Float16 v8h;
typedef __attribute__((ext_vector_type(8)))  float    v8f;
typedef __attribute__((ext_vector_type(2)))  float    f32x2;

#define NTILE 3   // column tiles per block

union HFrag { v16h v; _Float16 h[16]; };

// Load a float pair from an integer-selected address on the GLOBAL path
// (addrspace 1 -> global_load_b64, LOADcnt tracking only, no flat/DScnt).
__device__ __forceinline__ f32x2 load_g2(uint64_t addr) {
#if defined(__HIP_DEVICE_COMPILE__)
  return *(__attribute__((address_space(1))) const f32x2*)addr;
#else
  return *(const f32x2*)addr;
#endif
}

// K offset of fragment element-pair j (0..7) within a 32-wide K slice,
// per CDNA5 ISA 7.12.2 (16-bit A/B matrix layout, wave32).
__device__ __forceinline__ int kpair(int j, int half) {
  return ((j >> 2) << 4) + ((j & 3) << 1) + (half << 3);
}

// Ordered-int float atomics (buffers initialized to +/-INF).
__device__ __forceinline__ void atomicMinF(float* a, float v) {
  if (v >= 0.f) atomicMin((int*)a, __float_as_int(v));
  else          atomicMax((unsigned int*)a, __float_as_uint(v));
}
__device__ __forceinline__ void atomicMaxF(float* a, float v) {
  if (v >= 0.f) atomicMax((int*)a, __float_as_int(v));
  else          atomicMin((unsigned int*)a, __float_as_uint(v));
}

// f32 A fragment: contiguous f32 K-pairs, clamped + selected. K even.
__device__ __forceinline__ v16h load_a_frag(const float* __restrict__ Arow,
                                            int K, int k0, int half) {
  HFrag a;
#pragma unroll
  for (int j = 0; j < 8; ++j) {
    int  kb = k0 + kpair(j, half);
    bool ok = kb < K;                 // K even, kb even -> kb+1 < K too
    int  kc = ok ? kb : 0;
    f32x2 p = *(const f32x2*)(Arow + kc);
    a.h[2 * j]     = (_Float16)(ok ? p[0] : 0.f);
    a.h[2 * j + 1] = (_Float16)(ok ? p[1] : 0.f);
  }
  return a.v;
}

// f16 A fragment: two contiguous 16-byte runs per lane (global_load_b128).
// Requires lda % 8 == 0 and lda >= roundup8(K) (padded rows), K even.
__device__ __forceinline__ v16h load_a_frag(const _Float16* __restrict__ Arow,
                                            int K, int k0, int half) {
  HFrag a;
#pragma unroll
  for (int run = 0; run < 2; ++run) {
    int  kb = k0 + (run << 4) + (half << 3);   // multiple of 8
    int  kc = (kb < K) ? kb : 0;
    v8h v = *(const v8h*)(Arow + kc);
#pragma unroll
    for (int i = 0; i < 8; i += 2) {
      bool ok = (kb + i) < K;                  // per-pair select (K even)
      a.h[(run << 3) + i]     = ok ? v[i]     : (_Float16)0.f;
      a.h[(run << 3) + i + 1] = ok ? v[i + 1] : (_Float16)0.f;
    }
  }
  return a.v;
}

// ---------------------------------------------------------------------------
// Pack B [K,N] f32 (row-major, ldb) into f16 WMMA fragments:
//   Bp[(kt*ntiles_n + nt)*32 + lane][0..15]  (32 B contiguous per lane).
// ---------------------------------------------------------------------------
__global__ void __launch_bounds__(32)
pack_b_kernel(const float* __restrict__ B, _Float16* __restrict__ Bp,
              int K, int Nn, int ldb)
{
  const int lane = threadIdx.x, half = lane >> 4, l16 = lane & 15;
  const int kt = blockIdx.x, nt = blockIdx.y;
  const int col  = (nt << 4) + l16;
  const bool cok = col < Nn;
  const int colc = cok ? col : 0;
  HFrag b;
#pragma unroll
  for (int j = 0; j < 8; ++j) {
    int  kb = (kt << 5) + kpair(j, half);
    bool ok = cok && (kb < K);
    int  kc = (kb < K) ? kb : 0;
    float v0 = B[(size_t)kc * ldb + colc];
    float v1 = B[((size_t)kc + 1) * ldb + colc];
    b.h[2 * j]     = (_Float16)(ok ? v0 : 0.f);
    b.h[2 * j + 1] = (_Float16)(ok ? v1 : 0.f);
  }
  ((v16h*)Bp)[((size_t)kt * gridDim.y + nt) * 32 + lane] = b.v;
}

// ---------------------------------------------------------------------------
// Generic WMMA GEMM: C[M,N] = act(A[M,K] @ B + bias). B pre-packed.
// Each block: 16 rows x NTILE 16-col tiles. M multiple of 16.
// ACT: 0=none, 1=relu, 2=sigmoid (v_rcp fast path).
// ---------------------------------------------------------------------------
template <typename TA, typename TC, int ACT>
__global__ void __launch_bounds__(32)
wmma_gemm_t(const TA* __restrict__ A, const _Float16* __restrict__ Bp,
            const float* __restrict__ bias, TC* __restrict__ C,
            int Nn, int K, int lda, int ldc, int colOff, int ntn)
{
  const int lane = threadIdx.x, half = lane >> 4, l16 = lane & 15;
  const int tm = blockIdx.x << 4;
  const TA* Arow = A + (size_t)(tm + l16) * lda;
  const v16h* BpV = (const v16h*)Bp;

  int ti[NTILE];
#pragma unroll
  for (int i = 0; i < NTILE; ++i) {
    int t = blockIdx.y * NTILE + i;
    ti[i] = t < ntn ? t : ntn - 1;       // clamp overflow slots (store skipped)
  }

  v8f acc[NTILE] = {};
  int kt = 0;
  for (int k0 = 0; k0 < K; k0 += 32, ++kt) {
    v16h a = load_a_frag(Arow, K, k0, half);
    const v16h* Bk = BpV + (size_t)kt * ntn * 32 + lane;
#pragma unroll
    for (int i = 0; i < NTILE; ++i) {
      v16h b = Bk[(size_t)ti[i] * 32];
      acc[i] = __builtin_amdgcn_wmma_f32_16x16x32_f16(false, a, false, b,
                                                      (short)0, acc[i],
                                                      false, false);
    }
  }

#pragma unroll
  for (int i = 0; i < NTILE; ++i) {
    int t   = blockIdx.y * NTILE + i;
    int col = (t << 4) + l16;
    if (t < ntn && col < Nn) {
      float bb = bias[col];
#pragma unroll
      for (int r = 0; r < 8; ++r) {
        float v = acc[i][r] + bb;
        if (ACT == 1) v = v > 0.f ? v : 0.f;
        if (ACT == 2) v = __builtin_amdgcn_rcpf(1.f + __expf(-v));
        C[(size_t)(tm + r + (half << 3)) * ldc + colOff + col] = (TC)v;
      }
    }
  }
}

// ---------------------------------------------------------------------------
// Pre-NN WMMA GEMM with gathered A rows: A[e,:] = [x[dst[e]] | x[src[e]] | e].
// Region selection on integer addresses, loaded through addrspace(1) so it
// stays a global_load (flat cndmask address select, LOADcnt tracking only).
// Region boundaries (fi, 2fi) are even so pairs never straddle.
// K = 3*fi, N = fi.
// ---------------------------------------------------------------------------
__global__ void __launch_bounds__(32)
wmma_pre_gemm_kernel(const float* x, int fi,
                     const int* __restrict__ src, const int* __restrict__ dst,
                     const float* eenc,
                     const _Float16* __restrict__ Bp,
                     const float* __restrict__ bias,
                     float* __restrict__ C, int ldc, int colOff, int ntn)
{
  const int lane = threadIdx.x, half = lane >> 4, l16 = lane & 15;
  const int tm = blockIdx.x << 4;
  const int fi2 = fi * 2, K = fi * 3, Nn = fi;
  const int row = tm + l16;                 // always < E (E % 16 == 0)

  // Integer base addresses with region offsets pre-subtracted:
  //   addr(k) = base(region(k)) + 4*k
  const uint64_t bd = (uint64_t)(x + (size_t)dst[row] * fi);
  const uint64_t bs = (uint64_t)(x + (size_t)src[row] * fi) - (uint64_t)fi * 4u;
  const uint64_t be = (uint64_t)(eenc + (size_t)row * fi)   - (uint64_t)fi2 * 4u;
  const v16h* BpV = (const v16h*)Bp;

  int ti[NTILE];
#pragma unroll
  for (int i = 0; i < NTILE; ++i) {
    int t = blockIdx.y * NTILE + i;
    ti[i] = t < ntn ? t : ntn - 1;
  }

  v8f acc[NTILE] = {};
  int kt = 0;
  for (int k0 = 0; k0 < K; k0 += 32, ++kt) {
    HFrag a;
#pragma unroll
    for (int j = 0; j < 8; ++j) {
      int  kb = k0 + kpair(j, half);
      bool ok = kb < K;
      int  kc = ok ? kb : 0;
      uint64_t base = (kc < fi) ? bd : ((kc < fi2) ? bs : be);
      f32x2 v = load_g2(base + (uint64_t)kc * 4u);
      a.h[2 * j]     = (_Float16)(ok ? v[0] : 0.f);
      a.h[2 * j + 1] = (_Float16)(ok ? v[1] : 0.f);
    }
    const v16h* Bk = BpV + (size_t)kt * ntn * 32 + lane;
#pragma unroll
    for (int i = 0; i < NTILE; ++i) {
      v16h b = Bk[(size_t)ti[i] * 32];
      acc[i] = __builtin_amdgcn_wmma_f32_16x16x32_f16(false, a.v, false, b,
                                                      (short)0, acc[i],
                                                      false, false);
    }
  }

#pragma unroll
  for (int i = 0; i < NTILE; ++i) {
    int t   = blockIdx.y * NTILE + i;
    int col = (t << 4) + l16;
    if (t < ntn && col < Nn) {
      float bb = bias[col];
#pragma unroll
      for (int r = 0; r < 8; ++r)
        C[(size_t)(tm + r + (half << 3)) * ldc + colOff + col] = acc[i][r] + bb;
    }
  }
}

// ---------------------------------------------------------------------------
// Elementwise / scatter kernels
// ---------------------------------------------------------------------------
__global__ void fill_kernel(float* __restrict__ p, float v, size_t n) {
  size_t i = (size_t)blockIdx.x * blockDim.x + threadIdx.x;
  if (i < n) p[i] = v;
}

__global__ void degree_kernel(const int* __restrict__ dst,
                              float* __restrict__ cnt, int E) {
  int i = blockIdx.x * blockDim.x + threadIdx.x;
  if (i < E) atomicAdd(&cnt[dst[i]], 1.f);
}

__global__ void stats_kernel(const float* __restrict__ cnt,
                             float* __restrict__ stats, int Nn) {
  int i = blockIdx.x * blockDim.x + threadIdx.x;
  if (i < Nn) {
    float c = cnt[i];
    atomicAdd(&stats[0], c);                 // -> avg_lin * N
    atomicAdd(&stats[1], logf(c + 1.f));     // -> avg_log * N
  }
}

__global__ void scaler_kernel(const float* __restrict__ cnt,
                              const float* __restrict__ stats,
                              float* __restrict__ scal, int Nn) {
  int i = blockIdx.x * blockDim.x + threadIdx.x;
  if (i < Nn) {
    float avg_lin = stats[0] / (float)Nn;
    float avg_log = stats[1] / (float)Nn;
    float deg = cnt[i] < 1.f ? 1.f : cnt[i];
    float ld  = logf(deg + 1.f);
    scal[(size_t)i * 4 + 0] = ld / avg_log;      // amp
    scal[(size_t)i * 4 + 1] = avg_log / ld;      // att
    scal[(size_t)i * 4 + 2] = deg / avg_lin;     // lin
    scal[(size_t)i * 4 + 3] = avg_lin / deg;     // inv
  }
}

// m: [E, T*fi] -> per-node sum/sumsq/min/max [N, T*fi] via L2 atomics.
__global__ void scatter_agg_kernel(const float* __restrict__ m,
                                   const int* __restrict__ dst,
                                   float* __restrict__ s, float* __restrict__ q,
                                   float* __restrict__ mn, float* __restrict__ mx,
                                   int E, int Cc) {
  size_t tid = (size_t)blockIdx.x * blockDim.x + threadIdx.x;
  if (tid >= (size_t)E * Cc) return;
  int c = (int)(tid % Cc);
  int e = (int)(tid / Cc);
  float v = m[tid];
  size_t ai = (size_t)dst[e] * Cc + c;
  atomicAdd(&s[ai], v);
  atomicAdd(&q[ai], v * v);
  atomicMinF(&mn[ai], v);
  atomicMaxF(&mx[ai], v);
}

// Build P[t][n][C26p] (f16, padded rows) =
//   [x | agg | agg*amp | agg*att | agg*lin | agg*inv], agg on the fly.
__global__ void post_assemble_kernel(const float* __restrict__ x, int fi,
                                     const float* __restrict__ s,
                                     const float* __restrict__ q,
                                     const float* __restrict__ mn,
                                     const float* __restrict__ mx,
                                     const float* __restrict__ cnt,
                                     const float* __restrict__ scal,
                                     _Float16* __restrict__ P,
                                     int Nn, int C26p) {
  const int fi5 = 5 * fi;
  const int C26r = 26 * fi;                 // real row length
  size_t total = (size_t)2 * Nn * C26p;
  size_t tid = (size_t)blockIdx.x * blockDim.x + threadIdx.x;
  if (tid >= total) return;
  int    c = (int)(tid % C26p);
  size_t r = tid / C26p;
  int    n = (int)(r % Nn);
  int    t = (int)(r / Nn);

  float val = 0.f;
  if (c < C26r) {
    if (c < fi) {
      val = x[(size_t)n * fi + c];
    } else {
      int c1   = c - fi;
      int sIdx = c1 / fi5;          // 0=id, 1=amp, 2=att, 3=lin, 4=inv
      int c2   = c1 % fi5;
      int aIdx = c2 / fi;           // 0=sum, 1=mean, 2=min, 3=max, 4=std
      int f    = c2 % fi;
      size_t ai = ((size_t)n * 2 + t) * fi + f;   // agg layout [N, T, fi]
      float cn = cnt[n];
      float cc = cn < 1.f ? 1.f : cn;
      float sv = s[ai];
      float mean = sv / cc;
      float a;
      if      (aIdx == 0) a = sv;
      else if (aIdx == 1) a = mean;
      else if (aIdx == 2) a = (cn > 0.f) ? mn[ai] : 0.f;
      else if (aIdx == 3) a = (cn > 0.f) ? mx[ai] : 0.f;
      else {
        float ms = q[ai] / cc - mean * mean;
        a = sqrtf((ms > 0.f ? ms : 0.f) + 1e-5f);
      }
      float sc = (sIdx == 0) ? 1.f : scal[(size_t)n * 4 + (sIdx - 1)];
      val = a * sc;
    }
  }
  P[tid] = (_Float16)val;
}

// z(f16, stride 496) = cat(h3, h2, h1) -> [N, 490] (+zero pad)
__global__ void concat_kernel(const float* __restrict__ h3,
                              const float* __restrict__ h2,
                              const float* __restrict__ h1,
                              _Float16* __restrict__ z, int Nn) {
  size_t tid = (size_t)blockIdx.x * blockDim.x + threadIdx.x;
  if (tid >= (size_t)Nn * 496) return;
  int c = (int)(tid % 496);
  int n = (int)(tid / 496);
  float v = 0.f;
  if      (c < 280) v = h3[(size_t)n * 280 + c];
  else if (c < 420) v = h2[(size_t)n * 140 + (c - 280)];
  else if (c < 490) v = h1[(size_t)n * 70  + (c - 420)];
  z[tid] = (_Float16)v;
}

// ---------------------------------------------------------------------------
// Host orchestration
// ---------------------------------------------------------------------------
extern "C" void kernel_launch(void* const* d_in, const int* in_sizes, int n_in,
                              void* d_out, int out_size, void* d_ws, size_t ws_size,
                              hipStream_t stream)
{
  (void)in_sizes; (void)n_in; (void)out_size; (void)ws_size;

  const int Nn = 20000, E = 320000, ED = 22, T = 2;
  const int FI[3] = {32, 70, 140};
  const int FO[3] = {70, 140, 280};
  auto rup8  = [](int v) { return (v + 7) & ~7; };
  auto ntile = [](int n) { return (n + 15) / 16; };
  auto nblk  = [](int nt) { return (nt + NTILE - 1) / NTILE; };

  const float* x0        = (const float*)d_in[0];
  const int*   edge_idx  = (const int*)  d_in[1];
  const float* edge_attr = (const float*)d_in[2];
  const int* src = edge_idx;        // edge_index[0]
  const int* dst = edge_idx + E;    // edge_index[1]

  struct ConvP { const float *We,*Wlin,*Wpost,*Wpre,*be,*blin,*bpost,*bpre; };
  ConvP cp[3];
  for (int i = 0; i < 3; ++i) {
    int b = 3 + 8 * i;
    cp[i].We    = (const float*)d_in[b + 0];
    cp[i].Wlin  = (const float*)d_in[b + 1];
    cp[i].Wpost = (const float*)d_in[b + 2];
    cp[i].Wpre  = (const float*)d_in[b + 3];
    cp[i].be    = (const float*)d_in[b + 4];
    cp[i].blin  = (const float*)d_in[b + 5];
    cp[i].bpost = (const float*)d_in[b + 6];
    cp[i].bpre  = (const float*)d_in[b + 7];
  }
  const float *Wm[4], *bm[4];
  for (int i = 0; i < 4; ++i) {
    Wm[i] = (const float*)d_in[27 + 2 * i];
    bm[i] = (const float*)d_in[28 + 2 * i];
  }

  // -------- f32 workspace arena --------
  float* w = (float*)d_ws;
  size_t off = 0;
  auto alloc = [&](size_t n) { float* p = w + off; off += (n + 15) & ~(size_t)15; return p; };
  float* cnt   = alloc((size_t)Nn);
  float* stats = alloc(16);
  float* scal  = alloc((size_t)4 * Nn);
  float* h[3];
  for (int i = 0; i < 3; ++i) h[i] = alloc((size_t)Nn * FO[i]);
  float* eenc = alloc((size_t)E * 140 + 64);      // +pad for clamped K-tail reads
  float* m    = alloc((size_t)E * 280);
  float* aS   = alloc((size_t)Nn * 280);
  float* aQ   = alloc((size_t)Nn * 280);
  float* aMn  = alloc((size_t)Nn * 280);
  float* aMx  = alloc((size_t)Nn * 280);

  // -------- f16 arena (GEMM-only tensors, padded rows) --------
  _Float16* hw = (_Float16*)(w + off);
  size_t hoff = 0;
  auto hal = [&](size_t n) { _Float16* p = hw + hoff; hoff += (n + 31) & ~(size_t)31; return p; };
  _Float16* P  = hal((size_t)Nn * 2 * 3648);      // [T][N][rup8(26*fi)] max
  _Float16* t0 = hal((size_t)Nn * 280);           // [N][rup8(fo)] max
  _Float16* z  = hal((size_t)Nn * 496);
  _Float16* z1 = hal((size_t)Nn * 424);
  _Float16* z2 = hal((size_t)Nn * 184);
  _Float16* z3 = hal((size_t)Nn * 64);

  // packed-f16 weight arena
  _Float16* packArena = hw + hoff;
  size_t poff = 0;
  auto packB = [&](const float* B, int K, int Nv, int ldb) {
    int kt = (K + 31) / 32, nt = (Nv + 15) / 16;
    _Float16* p = packArena + poff;
    poff += (size_t)kt * nt * 512;
    pack_b_kernel<<<dim3(kt, nt), 32, 0, stream>>>(B, p, K, Nv, ldb);
    return p;
  };

  auto g1 = [](size_t n, int b) { return dim3((unsigned)((n + (size_t)b - 1) / b)); };

  // -------- degree statistics & scalers --------
  fill_kernel<<<g1(Nn, 256), 256, 0, stream>>>(cnt, 0.f, (size_t)Nn);
  fill_kernel<<<1, 256, 0, stream>>>(stats, 0.f, (size_t)16);
  degree_kernel<<<g1(E, 256), 256, 0, stream>>>(dst, cnt, E);
  stats_kernel<<<g1(Nn, 256), 256, 0, stream>>>(cnt, stats, Nn);
  scaler_kernel<<<g1(Nn, 256), 256, 0, stream>>>(cnt, stats, scal, Nn);

  // -------- 3 PNAConv layers --------
  const float* xin = x0;
  for (int L = 0; L < 3; ++L) {
    const int fi = FI[L], fo = FO[L], fot = fo / 2;
    const int K3 = 3 * fi, C26 = 26 * fi, C2 = T * fi;
    const int C26p = rup8(C26), fop = rup8(fo);
    const int ntFi = ntile(fi);

    // edge encoder: eenc = edge_attr @ We + be          [E, fi] (f32)
    {
      _Float16* pB = packB(cp[L].We, ED, fi, fi);
      wmma_gemm_t<float, float, 0>
          <<<dim3(E / 16, nblk(ntFi)), 32, 0, stream>>>(
          edge_attr, pB, cp[L].be, eenc, fi, ED, ED, fi, 0, ntFi);
    }

    // pre-NN per tower: m[:, t*fi:(t+1)*fi] = [x_i|x_j|e] @ Wpre[t] + bpre[t]
    for (int t = 0; t < T; ++t) {
      _Float16* pB = packB(cp[L].Wpre + (size_t)t * K3 * fi, K3, fi, fi);
      wmma_pre_gemm_kernel<<<dim3(E / 16, nblk(ntFi)), 32, 0, stream>>>(
          xin, fi, src, dst, eenc, pB, cp[L].bpre + t * fi, m, C2, t * fi, ntFi);
    }

    // segment sum/sumsq/min/max over dst
    size_t na = (size_t)Nn * C2;
    fill_kernel<<<g1(na, 256), 256, 0, stream>>>(aS, 0.f, na);
    fill_kernel<<<g1(na, 256), 256, 0, stream>>>(aQ, 0.f, na);
    fill_kernel<<<g1(na, 256), 256, 0, stream>>>(aMn,  INFINITY, na);
    fill_kernel<<<g1(na, 256), 256, 0, stream>>>(aMx, -INFINITY, na);
    size_t ne = (size_t)E * C2;
    scatter_agg_kernel<<<g1(ne, 256), 256, 0, stream>>>(m, dst, aS, aQ, aMn, aMx, E, C2);

    // assemble post-NN input [T][N][C26p] (f16) with scalers fused
    size_t np = (size_t)T * Nn * C26p;
    post_assemble_kernel<<<g1(np, 256), 256, 0, stream>>>(
        xin, fi, aS, aQ, aMn, aMx, cnt, scal, P, Nn, C26p);

    // post-NN per tower -> t0[N, fop] (f16, tower column windows)
    for (int t = 0; t < T; ++t) {
      _Float16* pB = packB(cp[L].Wpost + (size_t)t * C26 * fot, C26, fot, fot);
      int nt = ntile(fot);
      wmma_gemm_t<_Float16, _Float16, 0>
          <<<dim3(Nn / 16, nblk(nt)), 32, 0, stream>>>(
          P + (size_t)t * Nn * C26p, pB, cp[L].bpost + t * fot,
          t0, fot, C26, C26p, fop, t * fot, nt);
    }

    // final per-conv lin: h[L] = t0 @ Wlin + blin        [N, fo] (f32)
    {
      _Float16* pB = packB(cp[L].Wlin, fo, fo, fo);
      int nt = ntile(fo);
      wmma_gemm_t<_Float16, float, 0>
          <<<dim3(Nn / 16, nblk(nt)), 32, 0, stream>>>(
          t0, pB, cp[L].blin, h[L], fo, fo, fop, fo, 0, nt);
    }

    xin = h[L];
  }

  // -------- MLP head: cat(h3,h2,h1) -> 420 -> 180 -> 60 -> 1 --------
  concat_kernel<<<g1((size_t)Nn * 496, 256), 256, 0, stream>>>(h[2], h[1], h[0], z, Nn);
  {
    _Float16* pB = packB(Wm[0], 490, 420, 420);
    wmma_gemm_t<_Float16, _Float16, 1>
        <<<dim3(Nn / 16, nblk(27)), 32, 0, stream>>>(
        z, pB, bm[0], z1, 420, 490, 496, 424, 0, 27);
  }
  {
    _Float16* pB = packB(Wm[1], 420, 180, 180);
    wmma_gemm_t<_Float16, _Float16, 1>
        <<<dim3(Nn / 16, nblk(12)), 32, 0, stream>>>(
        z1, pB, bm[1], z2, 180, 420, 424, 184, 0, 12);
  }
  {
    _Float16* pB = packB(Wm[2], 180, 60, 60);
    wmma_gemm_t<_Float16, _Float16, 1>
        <<<dim3(Nn / 16, nblk(4)), 32, 0, stream>>>(
        z2, pB, bm[2], z3, 60, 180, 184, 64, 0, 4);
  }
  {
    _Float16* pB = packB(Wm[3], 60, 1, 1);
    wmma_gemm_t<_Float16, float, 2>
        <<<dim3(Nn / 16, 1), 32, 0, stream>>>(
        z3, pB, bm[3], (float*)d_out, 1, 60, 64, 1, 0, 1);
  }
}
